// AttentionHead_37623913513818
// MI455X (gfx1250) — compile-verified, matrix operand order checked by reading, not run on hardware
//
#include <hip/hip_runtime.h>

// ---------------------------------------------------------------------------
// Types for CDNA5 WMMA (wave32, 16x16x32 bf16 -> f32)
// ---------------------------------------------------------------------------
typedef __bf16 bf16;
typedef __attribute__((ext_vector_type(16))) __bf16 v16bf;
typedef __attribute__((ext_vector_type(8)))  __bf16 v8bf;
typedef __attribute__((ext_vector_type(4)))  __bf16 v4bf;
typedef __attribute__((ext_vector_type(8)))  float  v8f;

#define D_MODEL 1024
#define NSEQ    2048
#define BATCH   8
#define LDQK    2048   // q|k buffer row stride (q cols 0..1023, k cols 1024..2047)
#define KT      32     // keys per flash iteration
#define QT      32     // queries per workgroup (2 x 16-row subtiles)

__device__ __forceinline__ v8f wmma_bf16(v16bf a, v16bf b, v8f c) {
  // D = A(16x32 bf16) * B(32x16 bf16) + C(16x16 f32)
  return __builtin_amdgcn_wmma_f32_16x16x32_bf16(
      /*neg_a=*/false, a, /*neg_b=*/false, b,
      /*c_mod=*/(short)0, c, /*reuse_a=*/false, /*reuse_b=*/false);
}

// Load a 16x32 bf16 fragment from a row-major matrix [rows x ld].
// A-matrix layout (ISA 7.12.2, 16-bit A 16x32): lane m = lane&15, half = lane>>4,
// VGPR v<4 : K = half*8 + 2v,2v+1 ; VGPR v>=4 : K = 16 + half*8 + ...
// => per lane: two contiguous 16B loads at k0+half*8 and k0+half*8+16.
// Also valid as a B(32x16) fragment when the source is B^T row-major [n x k].
__device__ __forceinline__ v16bf load_frag_row16(const bf16* p, int ld, int row0, int k0) {
  int lane = threadIdx.x & 31;
  int m    = lane & 15;
  int half = lane >> 4;
  const bf16* r = p + (size_t)(row0 + m) * ld + k0 + half * 8;
  v8bf lo = *(const v8bf*)(r);
  v8bf hi = *(const v8bf*)(r + 16);
  return __builtin_shufflevector(lo, hi, 0, 1, 2, 3, 4, 5, 6, 7,
                                 8, 9, 10, 11, 12, 13, 14, 15);
}

// ---------------------------------------------------------------------------
// Kernel 1: fp32 -> bf16 (vectorized x4)
// ---------------------------------------------------------------------------
__global__ void __launch_bounds__(256) cvt_bf16_k(const float* __restrict__ in,
                                                  bf16* __restrict__ out, int n4) {
  int i = blockIdx.x * 256 + threadIdx.x;
  if (i < n4) {
    float4 v = ((const float4*)in)[i];
    v4bf r;
    r[0] = (bf16)v.x; r[1] = (bf16)v.y; r[2] = (bf16)v.z; r[3] = (bf16)v.w;
    ((v4bf*)out)[i] = r;
  }
}

// ---------------------------------------------------------------------------
// Kernel 2: QKV projection. val[m][o] = sum_c x[m][c]*W[o][c] + bias[o]
// Register-blocked: each wave owns a 64x64 output block (4x4 WMMA tiles).
// Per K-step: 4 A-frag + 4 B-frag loads feed 16 WMMAs (0.5 loads/WMMA),
// cutting L2 read traffic ~4x vs one-tile-per-wave.
// q,k go to qk[m][0..2047] (ld 2048). v goes out TRANSPOSED per batch:
// vT[b][d][n] (ld 2048) so the attention PV B-fragment becomes two b128 loads.
// ---------------------------------------------------------------------------
__global__ void __launch_bounds__(256) qkv_gemm_k(const bf16* __restrict__ xbf,
                                                  const bf16* __restrict__ wbf,
                                                  const float* __restrict__ bias,
                                                  bf16* __restrict__ qk,
                                                  bf16* __restrict__ vT) {
  int wave = threadIdx.x >> 5;
  int lane = threadIdx.x & 31;
  int blk  = blockIdx.x * 8 + wave;       // (16384/64)*(3072/64) = 12288 wave-blocks
  int tm = blk / 48;                      // 256 m-blocks of 64 rows
  int tn = blk - tm * 48;                 // 48  o-blocks of 64 cols
  int m0 = tm * 64, o0 = tn * 64;

  v8f acc[4][4];
#pragma unroll
  for (int i = 0; i < 4; ++i)
#pragma unroll
    for (int j = 0; j < 4; ++j) acc[i][j] = (v8f){};

  for (int c0 = 0; c0 < D_MODEL; c0 += 32) {
    v16bf a[4], b[4];
#pragma unroll
    for (int i = 0; i < 4; ++i)
      a[i] = load_frag_row16(xbf, D_MODEL, m0 + i * 16, c0);  // x rows
#pragma unroll
    for (int j = 0; j < 4; ++j)
      b[j] = load_frag_row16(wbf, D_MODEL, o0 + j * 16, c0);  // W is B^T row-major
#pragma unroll
    for (int i = 0; i < 4; ++i)
#pragma unroll
      for (int j = 0; j < 4; ++j)
        acc[i][j] = wmma_bf16(a[i], b[j], acc[i][j]);
  }

  int col = lane & 15, half = lane >> 4;
  if (o0 < 2048) {
    // q or k: row-major [m][o], ld 2048
#pragma unroll
    for (int j = 0; j < 4; ++j) {
      float bv = bias[o0 + j * 16 + col];
#pragma unroll
      for (int i = 0; i < 4; ++i) {
#pragma unroll
        for (int r = 0; r < 8; ++r) {
          int row = m0 + i * 16 + r + 8 * half;               // C layout: M = r + 8*half
          qk[(size_t)row * LDQK + o0 + j * 16 + col] = (bf16)(acc[i][j][r] + bv);
        }
      }
    }
  } else {
    // v: transposed store vT[b][d][n];  b = row/2048, n = row%2048, d = o-2048
#pragma unroll
    for (int j = 0; j < 4; ++j) {
      int d = o0 - 2048 + j * 16 + col;
      float bv = bias[o0 + j * 16 + col];
#pragma unroll
      for (int i = 0; i < 4; ++i) {
#pragma unroll
        for (int r = 0; r < 8; ++r) {
          int row = m0 + i * 16 + r + 8 * half;
          int b = row >> 11, n = row & 2047;
          vT[((size_t)b << 21) + ((size_t)d << 11) + n] = (bf16)(acc[i][j][r] + bv);
        }
      }
    }
  }
}

// ---------------------------------------------------------------------------
// Kernel 3: flash attention. 1 workgroup (8 waves, 256 thr) per 32-query block
// (2 x 16-row subtiles) -> every K/V fragment load feeds TWO WMMAs. Per
// 32-key iteration:
//   A: each wave, partial scores for both q-subtiles over its 128-wide D-slice
//      (ds_add_f32 reduction into padded LDS S)
//   B: softmax distributed across ALL 8 waves (4 rows each), exp2 domain
//   C: rescale accumulators, P*V (V fragments shared by both q-subtiles)
// ---------------------------------------------------------------------------
__global__ void __launch_bounds__(256) attn_k(const bf16* __restrict__ qk,
                                              const bf16* __restrict__ vT,
                                              float* __restrict__ out) {
  __shared__ __attribute__((aligned(16))) float S[QT][33];   // pad: halves hit diff banks
  __shared__ __attribute__((aligned(16))) bf16  P[QT * 32];  // row-major, ld 32
  __shared__ float Fac[QT];
  __shared__ float Linv[QT];

  int tid  = threadIdx.x;
  int wave = tid >> 5;
  int lane = tid & 31;
  int col  = lane & 15;
  int half = lane >> 4;

  int bIdx = blockIdx.x >> 6;              // / (NSEQ/32)
  int q0   = (blockIdx.x & 63) * QT;

  const bf16* qb  = qk + (size_t)bIdx * NSEQ * LDQK;         // q: cols [0,1024)
  const bf16* kb  = qb + 1024;                               // k: cols [1024,2048)
  const bf16* vtb = vT + ((size_t)bIdx << 21);               // vT: [1024][2048]

  int colw = wave * 128;                   // this wave's 128 output columns / D-slice

  v8f O[2][8];
#pragma unroll
  for (int qi = 0; qi < 2; ++qi)
#pragma unroll
    for (int t = 0; t < 8; ++t) O[qi][t] = (v8f){};

  // Q fragments for both q-subtiles over this wave's D-slice (loop-invariant)
  v16bf qfr[2][4];
#pragma unroll
  for (int qi = 0; qi < 2; ++qi)
#pragma unroll
    for (int cc = 0; cc < 4; ++cc)
      qfr[qi][cc] = load_frag_row16(qb, LDQK, q0 + qi * 16, colw + cc * 32);

  // Softmax ownership: wave w handles qi = w>>2, r in {2*(w&3), 2*(w&3)+1}
  int sqi = wave >> 2;
  int srb = (wave & 3) * 2;
  float m_s[2], l_s[2];
#pragma unroll
  for (int rr = 0; rr < 2; ++rr) { m_s[rr] = -3.0e38f; l_s[rr] = 0.0f; }

  for (int i = tid; i < QT * 33; i += 256) ((float*)S)[i] = 0.0f;
  __syncthreads();

  for (int j0 = 0; j0 < NSEQ; j0 += KT) {
    // Prefetch next iteration's K rows and V columns (global_prefetch_b8).
    if (j0 + KT < NSEQ) {
      const bf16* kpre = kb + (size_t)(j0 + KT + lane) * LDQK + colw;
      __builtin_prefetch(kpre, 0, 1);
      __builtin_prefetch(kpre + 64, 0, 1);
      const bf16* vpre = vtb + ((size_t)(colw + lane * 4) << 11) + j0 + KT;
#pragma unroll
      for (int rr = 0; rr < 4; ++rr)
        __builtin_prefetch(vpre + ((size_t)rr << 11), 0, 1);
    }

    // ---- Phase A: partial S for both q-subtiles over D-slice [colw, colw+128)
    v8f s00 = {}, s01 = {}, s10 = {}, s11 = {};
#pragma unroll
    for (int cc = 0; cc < 4; ++cc) {
      int c0 = colw + cc * 32;
      v16bf k0f = load_frag_row16(kb, LDQK, j0, c0);         // keys j0..j0+15
      v16bf k1f = load_frag_row16(kb, LDQK, j0 + 16, c0);    // keys j0+16..j0+31
      s00 = wmma_bf16(qfr[0][cc], k0f, s00);
      s01 = wmma_bf16(qfr[0][cc], k1f, s01);
      s10 = wmma_bf16(qfr[1][cc], k0f, s10);
      s11 = wmma_bf16(qfr[1][cc], k1f, s11);
    }
#pragma unroll
    for (int r = 0; r < 8; ++r) {
      int row = r + 8 * half;
      atomicAdd(&S[row][col],           s00[r]);             // ds_add_f32
      atomicAdd(&S[row][col + 16],      s01[r]);
      atomicAdd(&S[row + 16][col],      s10[r]);
      atomicAdd(&S[row + 16][col + 16], s11[r]);
    }
    __syncthreads();

    // ---- Phase B: distributed online softmax, exp2 domain.
    // scale = 1/sqrt(1024) * log2(e) folded into one constant.
    {
      const float sc2 = 0.03125f * 1.4426950408889634f;
#pragma unroll
      for (int rr = 0; rr < 2; ++rr) {
        int row = sqi * 16 + srb + rr + 8 * half;
        float a  = S[row][col] * sc2;
        float b2 = S[row][col + 16] * sc2;
        float mx = fmaxf(a, b2);
#pragma unroll
        for (int off = 1; off < 16; off <<= 1) mx = fmaxf(mx, __shfl_xor(mx, off));
        float mn = fmaxf(m_s[rr], mx);
        float p0 = exp2f(a - mn);
        float p1 = exp2f(b2 - mn);
        float rs = p0 + p1;
#pragma unroll
        for (int off = 1; off < 16; off <<= 1) rs += __shfl_xor(rs, off);
        float f = exp2f(m_s[rr] - mn);
        l_s[rr] = l_s[rr] * f + rs;
        m_s[rr] = mn;
        P[row * 32 + col]      = (bf16)p0;
        P[row * 32 + col + 16] = (bf16)p1;
        S[row][col]      = 0.0f;                              // re-arm for next iter
        S[row][col + 16] = 0.0f;
        if (col == 0) Fac[row] = f;
      }
    }
    __syncthreads();

    // ---- Phase C: rescale accumulators, then P*V (V fragment shared by both qi)
    float fr[2][8];
#pragma unroll
    for (int qi = 0; qi < 2; ++qi)
#pragma unroll
      for (int r = 0; r < 8; ++r) fr[qi][r] = Fac[qi * 16 + r + 8 * half];
#pragma unroll
    for (int qi = 0; qi < 2; ++qi)
#pragma unroll
      for (int t = 0; t < 8; ++t)
#pragma unroll
        for (int r = 0; r < 8; ++r) O[qi][t][r] *= fr[qi][r];

    v16bf pf0 = load_frag_row16((const bf16*)P, 32, 0, 0);    // probs rows 0..15
    v16bf pf1 = load_frag_row16((const bf16*)P + 16 * 32, 32, 0, 0); // rows 16..31
#pragma unroll
    for (int t = 0; t < 8; ++t) {
      // B[kk][n] = V[j0+kk][colw+16t+n] = vT[colw+16t+n][j0+kk]  (row-major [n][k])
      v16bf vf = load_frag_row16(vtb, NSEQ, colw + t * 16, j0);
      O[0][t] = wmma_bf16(pf0, vf, O[0][t]);
      O[1][t] = wmma_bf16(pf1, vf, O[1][t]);
    }
    // next-iteration Phase A barrier protects P/Fac reuse
  }

  if (col == 0) {
#pragma unroll
    for (int rr = 0; rr < 2; ++rr)
      Linv[sqi * 16 + srb + rr + 8 * half] = 1.0f / l_s[rr];
  }
  __syncthreads();

#pragma unroll
  for (int qi = 0; qi < 2; ++qi) {
#pragma unroll
    for (int r = 0; r < 8; ++r) {
      int row = qi * 16 + r + 8 * half;
      float li = Linv[row];
      size_t o = ((size_t)bIdx * NSEQ + q0 + row) * D_MODEL + colw + col;
#pragma unroll
      for (int t = 0; t < 8; ++t) out[o + t * 16] = O[qi][t][r] * li;
    }
  }
}

// ---------------------------------------------------------------------------
// Launch
// ---------------------------------------------------------------------------
extern "C" void kernel_launch(void* const* d_in, const int* in_sizes, int n_in,
                              void* d_out, int out_size, void* d_ws, size_t ws_size,
                              hipStream_t stream) {
  const float* x    = (const float*)d_in[0];  // [8,2048,1024]
  const float* Wq   = (const float*)d_in[1];  // [3072,1024]
  const float* bq   = (const float*)d_in[2];  // [3072]
  float*       out  = (float*)d_out;          // [8,2048,1024]

  const size_t W_ELT  = (size_t)3 * D_MODEL * D_MODEL;          // 3,145,728
  const size_t X_ELT  = (size_t)BATCH * NSEQ * D_MODEL;         // 16,777,216
  const size_t QK_ELT = (size_t)BATCH * NSEQ * LDQK;            // 33,554,432
  // vT: BATCH * D_MODEL * NSEQ = 16,777,216 elements

  char* ws  = (char*)d_ws;
  bf16* Wbf = (bf16*)ws;
  bf16* xbf = (bf16*)(ws + W_ELT * sizeof(bf16));
  bf16* qk  = (bf16*)(ws + (W_ELT + X_ELT) * sizeof(bf16));
  bf16* vT  = (bf16*)(ws + (W_ELT + X_ELT + QK_ELT) * sizeof(bf16));
  (void)ws_size; (void)in_sizes; (void)n_in; (void)out_size;

  // fp32 -> bf16 staging (one-time; lives in 192MB L2 afterwards)
  cvt_bf16_k<<<(int)(W_ELT / 4 / 256), 256, 0, stream>>>(Wq, Wbf, (int)(W_ELT / 4));
  cvt_bf16_k<<<(int)(X_ELT / 4 / 256), 256, 0, stream>>>(x,  xbf, (int)(X_ELT / 4));

  // QKV projection: 12288 64x64 wave-blocks / 8 waves per block
  qkv_gemm_k<<<12288 / 8, 256, 0, stream>>>(xbf, Wbf, bq, qk, vT);

  // Flash attention: one block per (batch, 32-query block)
  attn_k<<<BATCH * (NSEQ / QT), 256, 0, stream>>>(qk, vT, out);
}
